// Net_73624329388060
// MI455X (gfx1250) — compile-verified, matrix operand order checked by reading, not run on hardware
//
#include <hip/hip_runtime.h>
#include <hip/hip_bf16.h>
#include <math.h>

typedef __attribute__((ext_vector_type(16))) _Float16 v16h;
typedef __attribute__((ext_vector_type(8)))  float    v8f;
typedef __attribute__((ext_vector_type(2)))  float    v2f;
typedef __attribute__((ext_vector_type(4)))  int      i32x4;
typedef __attribute__((ext_vector_type(8)))  int      i32x8;

#define BATCH 32
#define NPTS  2048
#define KNN   4
#define NODES (BATCH * NPTS)   // 65536
#define EDGES (NODES * KNN)    // 262144

// -------- helpers ----------------------------------------------------------

// A/B operand K-index map for 16-bit 16x16x32 WMMA:
// lanes 0-15 hold K {0..7,16..23}; lanes 16-31 hold K {8..15,24..31}
__device__ __forceinline__ int a_kmap(int lane, int e) {
    return e + 8 * ((e >> 3) + ((lane >> 4) & 1));
}

// Branch-free top-4 (smallest) tracker kept entirely in VGPRs.
struct Top4 { float d0, d1, d2, d3; int i0, i1, i2, i3; };

__device__ __forceinline__ void top4_init(Top4& t) {
    t.d0 = t.d1 = t.d2 = t.d3 = 1e30f;
    t.i0 = t.i1 = t.i2 = t.i3 = 0;
}

__device__ __forceinline__ void top4_insert(float d, int j, Top4& t) {
    bool c0 = d < t.d0, c1 = d < t.d1, c2 = d < t.d2, c3 = d < t.d3;
    t.d3 = c3 ? (c2 ? t.d2 : d) : t.d3;  t.i3 = c3 ? (c2 ? t.i2 : j) : t.i3;
    t.d2 = c2 ? (c1 ? t.d1 : d) : t.d2;  t.i2 = c2 ? (c1 ? t.i1 : j) : t.i2;
    t.d1 = c1 ? (c0 ? t.d0 : d) : t.d1;  t.i1 = c1 ? (c0 ? t.i0 : j) : t.i1;
    t.d0 = c0 ? d : t.d0;                t.i0 = c0 ? j : t.i0;
}

__device__ __forceinline__ unsigned lds_off_of(const void* p) {
    // generic LDS address: low 32 bits are the LDS byte offset (ISA 10.2 aperture map)
    return (unsigned)(unsigned long long)(uintptr_t)p;
}

// Issue a 1-D contiguous TDM load of n f32 elements (n <= 16384) into LDS.
// D# per CDNA5 ISA: group0 {count=1, lds_addr,
// global_addr, type=2}; group1 {mask=0, data_size=4B, dims/tile/stride}.
__device__ __forceinline__ void tdm_load_1d_f32(unsigned lds_off, const void* g,
                                                unsigned n) {
    unsigned long long ga = (unsigned long long)(uintptr_t)g;
    i32x4 g0;
    g0[0] = 1;                                             // count=1, user mode
    g0[1] = (int)lds_off;                                  // lds_addr
    g0[2] = (int)(unsigned)(ga & 0xffffffffull);           // global_addr[31:0]
    g0[3] = (int)(((ga >> 32) & 0x1ffffffull) | 0x80000000u); // ga[56:32] | type=2
    i32x8 g1;
    g1[0] = 0x20000;                                       // data_size=2 (4 bytes)
    g1[1] = (int)((n & 0xffffu) << 16);                    // tensor_dim0[15:0]
    g1[2] = (int)((n >> 16) | (1u << 16));                 // tensor_dim0[31:16] | tensor_dim1=1
    g1[3] = (int)(n << 16);                                // tile_dim0 = n
    g1[4] = 1;                                             // tile_dim1=1, tile_dim2=0
    g1[5] = (int)n;                                        // tensor_dim0_stride
    g1[6] = 0;
    g1[7] = 0;
    asm volatile("tensor_load_to_lds %0, %1" :: "s"(g0), "s"(g1) : "memory");
}

// -------- kNN for 1-D features (conv1) -------------------------------------

__global__ __launch_bounds__(256)
void knn1_kernel(const float* __restrict__ x, int* __restrict__ idx) {
    __shared__ float xs[NPTS];
    int b   = blockIdx.x / (NPTS / 256);
    int seg = blockIdx.x % (NPTS / 256);
    const float* xb = x + (size_t)b * NPTS;
    for (int i = threadIdx.x; i < NPTS; i += 256) xs[i] = xb[i];
    __syncthreads();
    int i = seg * 256 + threadIdx.x;
    float xi = xs[i];
    Top4 t; top4_init(t);
    for (int j = 0; j < NPTS; ++j) {
        float dj = xs[j] - xi;
        top4_insert(dj * dj, j, t);
    }
    int base = (b * NPTS + i) * KNN;
    idx[base + 0] = t.i0; idx[base + 1] = t.i1;
    idx[base + 2] = t.i2; idx[base + 3] = t.i3;
}

// -------- squared norms ----------------------------------------------------

__global__ __launch_bounds__(256)
void sqnorm_kernel(const float* __restrict__ f, int d, float* __restrict__ sq) {
    int n = blockIdx.x * 256 + threadIdx.x;
    const float* r = f + (size_t)n * d;
    float s = 0.f;
    for (int c = 0; c < d; ++c) { float v = r[c]; s += v * v; }
    sq[n] = s;
}

// -------- kNN via f32 WMMA Gram matrix (d == 32) ---------------------------
// grid: BATCH * (NPTS/128), block 256 (8 waves); each wave owns 16 query rows.
// G-tile handoff is wave-local (per-wave LDS region + in-order LDS pipe),
// so no block barriers in the column loop.

__global__ __launch_bounds__(256)
void knn_gram_kernel(const float* __restrict__ feat, const float* __restrict__ sq,
                     int* __restrict__ idx) {
    __shared__ float gtile[8][16][16];
    int b     = blockIdx.x / (NPTS / 128);
    int strip = blockIdx.x % (NPTS / 128);
    int wave  = threadIdx.x >> 5;
    int lane  = threadIdx.x & 31;
    int row   = strip * 128 + wave * 16 + (lane & 15);
    const float* fb  = feat + (size_t)b * NPTS * 32;
    const float* sqb = sq + (size_t)b * NPTS;

    // preload this lane's A slice: 8 k-steps of v2f (f32 16x16x4 layout)
    float areg[16];
    {
        int koff = ((lane >> 4) & 1) * 2;
        const float* ar = fb + (size_t)row * 32;
#pragma unroll
        for (int s = 0; s < 8; ++s) {
            areg[2 * s]     = ar[4 * s + koff];
            areg[2 * s + 1] = ar[4 * s + koff + 1];
        }
    }
    float my_sqi = sqb[row];
    Top4 t; top4_init(t);
    int rhalf = (lane >> 4) & 1;

    for (int jt = 0; jt < NPTS / 16; ++jt) {
        int col = jt * 16 + (lane & 15);
        const float* br = fb + (size_t)col * 32;
        int koff = rhalf * 2;
        v8f c = {};
#pragma unroll
        for (int s = 0; s < 8; ++s) {
            v2f av, bv;
            av[0] = areg[2 * s]; av[1] = areg[2 * s + 1];
            bv[0] = br[4 * s + koff]; bv[1] = br[4 * s + koff + 1];
            c = __builtin_amdgcn_wmma_f32_16x16x4_f32(false, av, false, bv,
                                                      (short)0, c, false, false);
        }
#pragma unroll
        for (int r = 0; r < 8; ++r) gtile[wave][r + 8 * rhalf][lane & 15] = c[r];
        asm volatile("s_wait_dscnt 0x0" ::: "memory");
        if (lane < 16) {
#pragma unroll
            for (int cc = 0; cc < 16; ++cc) {
                int j = jt * 16 + cc;
                float d2 = my_sqi + sqb[j] - 2.f * gtile[wave][lane][cc];
                top4_insert(d2, j, t);
            }
        }
    }
    if (lane < 16) {
        int base = (b * NPTS + row) * KNN;
        idx[base + 0] = t.i0; idx[base + 1] = t.i1;
        idx[base + 2] = t.i2; idx[base + 3] = t.i3;
    }
}

// -------- edge MLP lin1 (+ BN statistics) ----------------------------------
// edge e -> node i = e/4; features [x_i | x_j - x_i], K padded to mult of 32.
// grid EDGES/128, block 256; 16 edges per wave. f16 WMMA, f32 accumulate.

__global__ __launch_bounds__(256)
void edge_lin1_kernel(const float* __restrict__ feat, const int* __restrict__ idx,
                      const float* __restrict__ W, const float* __restrict__ bias,
                      int din, int cout,
                      float* __restrict__ h, float* __restrict__ gsum,
                      float* __restrict__ gsq) {
    __shared__ _Float16 atile[128][64];
    __shared__ float lsum[64], lsq[64];
    int Kpad = (2 * din + 31) & ~31;
    int Kact = 2 * din;

    for (int t = threadIdx.x; t < 128 * Kpad; t += 256) {
        int e_local = t / Kpad, c = t % Kpad;
        int e = blockIdx.x * 128 + e_local;
        int node = e >> 2;
        int b = node >> 11;          // NPTS = 2048
        int j = idx[e];
        const float* fi = feat + (size_t)node * din;
        const float* fj = feat + ((size_t)(b << 11) + j) * din;
        float v = 0.f;
        if (c < din) v = fi[c];
        else if (c < Kact) v = fj[c - din] - fi[c - din];
        atile[e_local][c] = (_Float16)v;
    }
    for (int t = threadIdx.x; t < cout; t += 256) { lsum[t] = 0.f; lsq[t] = 0.f; }
    __syncthreads();

    int wave = threadIdx.x >> 5, lane = threadIdx.x & 31;
    int erow0 = wave * 16;
    int ebase = blockIdx.x * 128 + erow0;
    int ncol = lane & 15;
    int rhalf = (lane >> 4) & 1;

    for (int nt = 0; nt < cout / 16; ++nt) {
        int col = nt * 16 + ncol;
        float bcol = bias[col];
        v8f c;
#pragma unroll
        for (int r = 0; r < 8; ++r) c[r] = bcol;
        for (int kt = 0; kt < Kpad / 32; ++kt) {
            v16h a, bm;
#pragma unroll
            for (int e = 0; e < 16; ++e) {
                int k = kt * 32 + a_kmap(lane, e);
                a[e]  = atile[erow0 + (lane & 15)][k];
                bm[e] = (k < Kact) ? (_Float16)W[(size_t)k * cout + col] : (_Float16)0.f;
            }
            c = __builtin_amdgcn_wmma_f32_16x16x32_f16(false, a, false, bm,
                                                       (short)0, c, false, false);
        }
        float ps = 0.f, pq = 0.f;
#pragma unroll
        for (int r = 0; r < 8; ++r) {
            float v = c[r];
            h[(size_t)(ebase + r + 8 * rhalf) * cout + col] = v;
            ps += v; pq += v * v;
        }
        atomicAdd(&lsum[col], ps);
        atomicAdd(&lsq[col], pq);
    }
    __syncthreads();
    for (int t = threadIdx.x; t < cout; t += 256) {
        atomicAdd(&gsum[t], lsum[t]);
        atomicAdd(&gsq[t], lsq[t]);
    }
}

// -------- fold BN stats into per-channel scale/shift -----------------------

__global__ void bn_finalize_kernel(const float* __restrict__ gsum, const float* __restrict__ gsq,
                                   const float* __restrict__ gamma, const float* __restrict__ beta,
                                   float invM, int cout,
                                   float* __restrict__ scale, float* __restrict__ shift) {
    int c = threadIdx.x;
    if (c < cout) {
        float mu  = gsum[c] * invM;
        float var = gsq[c] * invM - mu * mu;
        float sc  = gamma[c] * rsqrtf(var + 1e-5f);
        scale[c] = sc;
        shift[c] = beta[c] - mu * sc;
    }
}

// -------- BN + ReLU + lin2 + max over k=4 neighbors ------------------------
// grid NODES/32, block 256; wave -> 16 edges = 4 nodes.
// The 128 x cout slab of h is contiguous -> staged into LDS by the TDM.

__global__ __launch_bounds__(256)
void edge_lin2max_kernel(const float* __restrict__ h, const float* __restrict__ scale,
                         const float* __restrict__ shift, const float* __restrict__ W,
                         const float* __restrict__ bias, int cout,
                         float* __restrict__ out) {
    __shared__ float    hs[128 * 64];
    __shared__ _Float16 atile[128][64];

    if ((threadIdx.x >> 5) == 0) {     // wave 0 issues the tensor DMA
        unsigned n = (unsigned)(128 * cout);
        tdm_load_1d_f32(lds_off_of(hs), h + (size_t)blockIdx.x * 128 * cout, n);
        __builtin_amdgcn_s_wait_tensorcnt(0);
    }
    __syncthreads();

    for (int t = threadIdx.x; t < 128 * cout; t += 256) {
        int c = t % cout;
        float v = hs[t] * scale[c] + shift[c];
        atile[t / cout][c] = (_Float16)fmaxf(v, 0.f);
    }
    __syncthreads();

    int wave = threadIdx.x >> 5, lane = threadIdx.x & 31;
    int erow0 = wave * 16;
    int ncol = lane & 15;
    int rhalf = (lane >> 4) & 1;
    size_t node0 = (size_t)blockIdx.x * 32 + wave * 4;

    for (int nt = 0; nt < cout / 16; ++nt) {
        int col = nt * 16 + ncol;
        float bcol = bias[col];
        v8f c;
#pragma unroll
        for (int r = 0; r < 8; ++r) c[r] = bcol;
        for (int kt = 0; kt < cout / 32; ++kt) {
            v16h a, bm;
#pragma unroll
            for (int e = 0; e < 16; ++e) {
                int k = kt * 32 + a_kmap(lane, e);
                a[e]  = atile[erow0 + (lane & 15)][k];
                bm[e] = (_Float16)W[(size_t)k * cout + col];
            }
            c = __builtin_amdgcn_wmma_f32_16x16x32_f16(false, a, false, bm,
                                                       (short)0, c, false, false);
        }
        // rows 0-3 -> node0, 4-7 -> node1 (low lanes); 8-11 -> node2, 12-15 -> node3
        float m01 = fmaxf(fmaxf(c[0], c[1]), fmaxf(c[2], c[3]));
        float m23 = fmaxf(fmaxf(c[4], c[5]), fmaxf(c[6], c[7]));
        out[(node0 + rhalf * 2 + 0) * cout + col] = m01;
        out[(node0 + rhalf * 2 + 1) * cout + col] = m23;
    }
}

// -------- head MLP GEMM ----------------------------------------------------
// grid NODES/64, block 128 (4 waves); 16 rows per wave, loops all N tiles.
// When A != nullptr the 64 x Kdim A-slab is contiguous -> TDM-staged in LDS.
// When A == nullptr, gathers the x1|x2|x3 concat directly from global.

__global__ __launch_bounds__(128)
void head_gemm_kernel(const float* __restrict__ A, const float* __restrict__ x1,
                      const float* __restrict__ x2, const float* __restrict__ x3,
                      const float* __restrict__ W, const float* __restrict__ bias,
                      int Kdim, int Ncols, int relu, float* __restrict__ out) {
    __shared__ float As[64 * 256];     // 64 KB
    int wave = threadIdx.x >> 5, lane = threadIdx.x & 31;
    int rowbase = blockIdx.x * 64 + wave * 16;
    int row = rowbase + (lane & 15);
    int ncol = lane & 15;
    int rhalf = (lane >> 4) & 1;
    int ktiles = Kdim / 32;

    if (A) {
        if (wave == 0) {
            tdm_load_1d_f32(lds_off_of(As),
                            A + (size_t)blockIdx.x * 64 * Kdim,
                            (unsigned)(64 * Kdim));
            __builtin_amdgcn_s_wait_tensorcnt(0);
        }
        __syncthreads();
    }

    v16h areg[8];
    for (int kt = 0; kt < ktiles; ++kt) {
#pragma unroll
        for (int e = 0; e < 16; ++e) {
            int k = kt * 32 + a_kmap(lane, e);
            float v;
            if (A) v = As[(size_t)(wave * 16 + (lane & 15)) * Kdim + k];
            else if (k < 32)      v = x1[(size_t)row * 32 + k];
            else if (k < 64)      v = x2[(size_t)row * 32 + (k - 32)];
            else                  v = x3[(size_t)row * 64 + (k - 64)];
            areg[kt][e] = (_Float16)v;
        }
    }
    for (int nt = 0; nt < Ncols / 16; ++nt) {
        int col = nt * 16 + ncol;
        float bcol = bias[col];
        v8f c;
#pragma unroll
        for (int r = 0; r < 8; ++r) c[r] = bcol;
        for (int kt = 0; kt < ktiles; ++kt) {
            v16h bm;
#pragma unroll
            for (int e = 0; e < 16; ++e) {
                int k = kt * 32 + a_kmap(lane, e);
                bm[e] = (_Float16)W[(size_t)k * Ncols + col];
            }
            c = __builtin_amdgcn_wmma_f32_16x16x32_f16(false, areg[kt], false, bm,
                                                       (short)0, c, false, false);
        }
#pragma unroll
        for (int r = 0; r < 8; ++r) {
            float v = c[r];
            if (relu) v = fmaxf(v, 0.f);
            out[(size_t)(rowbase + r + 8 * rhalf) * Ncols + col] = v;
        }
    }
}

// -------- final 256 -> 2 layer + log_softmax -------------------------------

__global__ __launch_bounds__(256)
void final_logsoftmax_kernel(const float* __restrict__ Hin, const float* __restrict__ W,
                             const float* __restrict__ bias, float* __restrict__ out) {
    int row = blockIdx.x * 256 + threadIdx.x;
    const float* hr = Hin + (size_t)row * 256;
    float z0 = bias[0], z1 = bias[1];
    for (int k = 0; k < 256; ++k) {
        float v = hr[k];
        z0 += v * W[k * 2];
        z1 += v * W[k * 2 + 1];
    }
    float m = fmaxf(z0, z1);
    float l = m + logf(expf(z0 - m) + expf(z1 - m));
    out[(size_t)row * 2]     = z0 - l;
    out[(size_t)row * 2 + 1] = z1 - l;
}

__global__ void zero_kernel(float* p, int n) {
    int i = blockIdx.x * 256 + threadIdx.x;
    if (i < n) p[i] = 0.f;
}

// -------- host orchestration ----------------------------------------------

extern "C" void kernel_launch(void* const* d_in, const int* in_sizes, int n_in,
                              void* d_out, int out_size, void* d_ws, size_t ws_size,
                              hipStream_t stream) {
    const float* x = (const float*)d_in[0];
    // conv params: lin1.W, lin1.b, gamma, beta, lin2.W, lin2.b
    const float* c1W1 = (const float*)d_in[1];  const float* c1b1 = (const float*)d_in[2];
    const float* c1g  = (const float*)d_in[3];  const float* c1be = (const float*)d_in[4];
    const float* c1W2 = (const float*)d_in[5];  const float* c1b2 = (const float*)d_in[6];
    const float* c2W1 = (const float*)d_in[7];  const float* c2b1 = (const float*)d_in[8];
    const float* c2g  = (const float*)d_in[9];  const float* c2be = (const float*)d_in[10];
    const float* c2W2 = (const float*)d_in[11]; const float* c2b2 = (const float*)d_in[12];
    const float* c3W1 = (const float*)d_in[13]; const float* c3b1 = (const float*)d_in[14];
    const float* c3g  = (const float*)d_in[15]; const float* c3be = (const float*)d_in[16];
    const float* c3W2 = (const float*)d_in[17]; const float* c3b2 = (const float*)d_in[18];
    const float* m0W = (const float*)d_in[19]; const float* m0b = (const float*)d_in[20];
    const float* m1W = (const float*)d_in[21]; const float* m1b = (const float*)d_in[22];
    const float* m2W = (const float*)d_in[23]; const float* m2b = (const float*)d_in[24];
    const float* m3W = (const float*)d_in[25]; const float* m3b = (const float*)d_in[26];

    char* wsb = (char*)d_ws;
    float* x1   = (float*)(wsb + (size_t)0);          //  8 MB [NODES,32]
    float* x2   = (float*)(wsb + ((size_t)8  << 20)); //  8 MB [NODES,32]
    float* x3   = (float*)(wsb + ((size_t)16 << 20)); // 16 MB [NODES,64]
    int*   knn  = (int*)  (wsb + ((size_t)32 << 20)); //  1 MB [NODES,4]
    float* sq   = (float*)(wsb + ((size_t)33 << 20)); // [NODES]
    float* st   = (float*)(wsb + ((size_t)34 << 20)); // stats: sum|sumsq|scale|shift
    float* gsum = st, *gsq = st + 64, *scl = st + 128, *shf = st + 192;
    float* hbuf = (float*)(wsb + ((size_t)35 << 20)); // 64 MB [EDGES,64] (reused)
    float* hid1 = (float*)(wsb + ((size_t)99 << 20)); // 64 MB [NODES,256]

    const float invM = 1.f / (float)EDGES;

    // ---- conv1 (din=1, cout=32) ----
    knn1_kernel<<<BATCH * (NPTS / 256), 256, 0, stream>>>(x, knn);
    zero_kernel<<<1, 256, 0, stream>>>(st, 128);
    edge_lin1_kernel<<<EDGES / 128, 256, 0, stream>>>(x, knn, c1W1, c1b1, 1, 32, hbuf, gsum, gsq);
    bn_finalize_kernel<<<1, 64, 0, stream>>>(gsum, gsq, c1g, c1be, invM, 32, scl, shf);
    edge_lin2max_kernel<<<NODES / 32, 256, 0, stream>>>(hbuf, scl, shf, c1W2, c1b2, 32, x1);

    // ---- conv2 (din=32, cout=32) ----
    sqnorm_kernel<<<NODES / 256, 256, 0, stream>>>(x1, 32, sq);
    knn_gram_kernel<<<BATCH * (NPTS / 128), 256, 0, stream>>>(x1, sq, knn);
    zero_kernel<<<1, 256, 0, stream>>>(st, 128);
    edge_lin1_kernel<<<EDGES / 128, 256, 0, stream>>>(x1, knn, c2W1, c2b1, 32, 32, hbuf, gsum, gsq);
    bn_finalize_kernel<<<1, 64, 0, stream>>>(gsum, gsq, c2g, c2be, invM, 32, scl, shf);
    edge_lin2max_kernel<<<NODES / 32, 256, 0, stream>>>(hbuf, scl, shf, c2W2, c2b2, 32, x2);

    // ---- conv3 (din=32, cout=64) ----
    sqnorm_kernel<<<NODES / 256, 256, 0, stream>>>(x2, 32, sq);
    knn_gram_kernel<<<BATCH * (NPTS / 128), 256, 0, stream>>>(x2, sq, knn);
    zero_kernel<<<1, 256, 0, stream>>>(st, 128);
    edge_lin1_kernel<<<EDGES / 128, 256, 0, stream>>>(x2, knn, c3W1, c3b1, 32, 64, hbuf, gsum, gsq);
    bn_finalize_kernel<<<1, 64, 0, stream>>>(gsum, gsq, c3g, c3be, invM, 64, scl, shf);
    edge_lin2max_kernel<<<NODES / 32, 256, 0, stream>>>(hbuf, scl, shf, c3W2, c3b2, 64, x3);

    // ---- head MLP: concat(128) -> 256 -> 256 -> 256 -> 2 + log_softmax ----
    head_gemm_kernel<<<NODES / 64, 128, 0, stream>>>(nullptr, x1, x2, x3, m0W, m0b,
                                                     128, 256, 1, hid1);
    head_gemm_kernel<<<NODES / 64, 128, 0, stream>>>(hid1, nullptr, nullptr, nullptr, m1W, m1b,
                                                     256, 256, 1, hbuf);
    head_gemm_kernel<<<NODES / 64, 128, 0, stream>>>(hbuf, nullptr, nullptr, nullptr, m2W, m2b,
                                                     256, 256, 1, hid1);
    final_logsoftmax_kernel<<<NODES / 256, 256, 0, stream>>>(hid1, m3W, m3b, (float*)d_out);
}
